// RelationAwareAttentionHead_45947560132770
// MI455X (gfx1250) — compile-verified, matrix operand order checked by reading, not run on hardware
//
#include <hip/hip_runtime.h>
#include <hip/hip_bf16.h>

// Problem constants (from the reference)
#define B_ 8
#define L_ 1024
#define H_ 1024
#define D_ 64
#define QT 16            // query rows per workgroup
#define KSPLIT 4         // k-range splits (flash partials merged in reduce)
#define KT 32            // key columns per inner tile
#define KRANGE (L_ / KSPLIT)   // 256
#define NTILES (KRANGE / KT)   // 8

typedef __attribute__((ext_vector_type(16))) __bf16 v16bf;
typedef __attribute__((ext_vector_type(8)))  __bf16 v8bf;
typedef __attribute__((ext_vector_type(8)))  float  v8f;

// Element->K mapping for 16-bit A/B fragments (wave32):
// lane half h: e<8 -> K = h*8+e ; e>=8 -> K = 16 + h*8 + (e-8)
// => each 8-element half of a fragment is CONTIGUOUS in memory.
__device__ __forceinline__ int kmap(int h, int e) {
    return (e < 8) ? (h * 8 + e) : (16 + h * 8 + (e - 8));
}

__device__ __forceinline__ v8f wmma_bf16(v16bf a, v16bf b, v8f c) {
    return __builtin_amdgcn_wmma_f32_16x16x32_bf16(
        false, a, false, b, (short)0, c, false, false);
}

// One 16-B load per fragment half (global or LDS, bf16 source).
__device__ __forceinline__ void frag_from_bf16(v16bf& f, const __bf16* base, int h) {
    const v8bf lo = *(const v8bf*)(base + h * 8);
    const v8bf hi = *(const v8bf*)(base + 16 + h * 8);
#pragma unroll
    for (int e = 0; e < 8; ++e) { f[e] = lo[e]; f[8 + e] = hi[e]; }
}

// Two float4 loads per fragment half (f32 source, convert to bf16).
__device__ __forceinline__ void frag_from_f32(v16bf& f, const float* base, int h) {
    const float4 a0 = *(const float4*)(base + h * 8);
    const float4 a1 = *(const float4*)(base + h * 8 + 4);
    const float4 b0 = *(const float4*)(base + 16 + h * 8);
    const float4 b1 = *(const float4*)(base + 16 + h * 8 + 4);
    f[0] = (__bf16)a0.x; f[1] = (__bf16)a0.y; f[2]  = (__bf16)a0.z; f[3]  = (__bf16)a0.w;
    f[4] = (__bf16)a1.x; f[5] = (__bf16)a1.y; f[6]  = (__bf16)a1.z; f[7]  = (__bf16)a1.w;
    f[8] = (__bf16)b0.x; f[9] = (__bf16)b0.y; f[10] = (__bf16)b0.z; f[11] = (__bf16)b0.w;
    f[12] = (__bf16)b1.x; f[13] = (__bf16)b1.y; f[14] = (__bf16)b1.z; f[15] = (__bf16)b1.w;
}

__device__ __forceinline__ void frag_zero(v16bf& f) {
#pragma unroll
    for (int e = 0; e < 16; ++e) f[e] = (__bf16)0.0f;
}

// ---------------------------------------------------------------------------
// Kernel 0: transpose+convert weights -> Wt[3][D][H] bf16 (contiguous K reads
// for the projection B-fragments). Tiny: 3*64*1024 elements.
// ---------------------------------------------------------------------------
__global__ __launch_bounds__(256) void wt_kernel(
    const float* __restrict__ Wq, const float* __restrict__ Wk,
    const float* __restrict__ Wv, __bf16* __restrict__ Wt)
{
    const int sel = blockIdx.x / D_;
    const int col = blockIdx.x % D_;
    const float* W = (sel == 0) ? Wq : (sel == 1) ? Wk : Wv;
    __bf16* dst = Wt + ((size_t)sel * D_ + col) * H_;
    for (int k = threadIdx.x; k < H_; k += 256)
        dst[k] = (__bf16)W[(size_t)k * D_ + col];
}

// ---------------------------------------------------------------------------
// Kernel 1: q/k/v projections  [8192x1024]f32 @ [1024x64] -> bf16 [8192x64].
// Q,K stored row-major; V stored TRANSPOSED Vt[b][d][L] so that the
// attention values_1 B-fragments become contiguous 16-B loads.
// ---------------------------------------------------------------------------
__global__ __launch_bounds__(32) void proj_kernel(
    const float* __restrict__ query, const float* __restrict__ key,
    const float* __restrict__ value,
    const __bf16* __restrict__ Wt,
    const float* __restrict__ bq, const float* __restrict__ bk,
    const float* __restrict__ bv,
    __bf16* __restrict__ Qbf, __bf16* __restrict__ Kbf, __bf16* __restrict__ Vt)
{
    const int lane = threadIdx.x & 31;
    const int h    = lane >> 4;
    const int nn   = lane & 15;
    const int mt   = blockIdx.x;   // 0..511 row tile over B*L
    const int nt   = blockIdx.y;   // 0..3  col tile over D
    const int sel  = blockIdx.z;   // 0:q 1:k 2:v

    const float* X    = (sel == 0) ? query : (sel == 1) ? key : value;
    const float* bias = (sel == 0) ? bq    : (sel == 1) ? bk  : bv;

    const int row = mt * 16 + nn;   // A-fragment row for this lane
    const int col = nt * 16 + nn;   // B-fragment column for this lane
    const __bf16* wb = Wt + ((size_t)sel * D_ + col) * H_;

    v8f acc = {};
    for (int kk = 0; kk < H_; kk += 32) {
        v16bf a, b;
        frag_from_f32(a, X + (size_t)row * H_ + kk, h);
        frag_from_bf16(b, wb + kk, h);
        acc = wmma_bf16(a, b, acc);
    }
    const float bb = bias[col];

    if (sel == 2) {
        // Transposed store: the 8 C-rows of this lane are contiguous in L.
        v8bf pk;
#pragma unroll
        for (int r = 0; r < 8; ++r) pk[r] = (__bf16)(acc[r] + bb);
        const int m0 = mt * 16 + h * 8;       // first global row of this half
        const int b  = m0 >> 10;
        const int l  = m0 & (L_ - 1);
        *(v8bf*)(Vt + ((size_t)b * D_ + col) * L_ + l) = pk;
    } else {
        __bf16* OUT = (sel == 0) ? Qbf : Kbf;
#pragma unroll
        for (int r = 0; r < 8; ++r) {
            const int m = mt * 16 + r + h * 8;
            OUT[(size_t)m * D_ + col] = (__bf16)(acc[r] + bb);
        }
    }
}

// ---------------------------------------------------------------------------
// Kernel 2: fused relation-aware flash attention (partial over k-split).
// grid = (64 q-tiles, 4 k-splits); block = 256 (8 waves).
//   wave w == batch w for attn_1 / values_1
//   wave w == q-rows {2w, 2w+1} for attn_2 / values_2 (batches as WMMA rows)
// Bias matrices are streamed exactly once per q-tile (shared over all B=8).
// ---------------------------------------------------------------------------
__global__ __launch_bounds__(256) void attn_kernel(
    const __bf16* __restrict__ Qbf, const __bf16* __restrict__ Kbf,
    const __bf16* __restrict__ Vt,
    const float* __restrict__ kbias, const float* __restrict__ vbias,
    const int* __restrict__ mask,
    float* __restrict__ O1p, float* __restrict__ O2p, float* __restrict__ stats)
{
    __shared__ __align__(16) __bf16 sQ[B_][QT][D_];   // 16 KB
    __shared__ float  sS2[QT][B_][KT];                // 16 KB
    __shared__ __align__(16) __bf16 sP[B_][QT][KT];   //  8 KB
    __shared__ float  sScale[B_][QT];                 // .5 KB

    const int lane = threadIdx.x & 31;
    const int w    = threadIdx.x >> 5;      // wave id == batch id
    const int h    = lane >> 4;
    const int nn   = lane & 15;
    const int qt   = blockIdx.x;
    const int ks   = blockIdx.y;
    const int q0   = qt * QT;
    const int k0   = ks * KRANGE;

    // Stage this q-tile (all 8 batches) into LDS.
    {
        const uint4* s = (const uint4*)(Qbf + ((size_t)w * L_ + q0) * D_);
        uint4* d = (uint4*)&sQ[w][0][0];
        for (int i = lane; i < (QT * D_ * 2) / 16; i += 32) d[i] = s[i];
    }
    __syncthreads();

    // A1: Q[batch=w] 16x64 (two K=32 chunks), resident for the whole loop.
    v16bf a1[2];
#pragma unroll
    for (int c = 0; c < 2; ++c)
        frag_from_bf16(a1[c], &sQ[w][nn][c * 32], h);

    // A2: rows = batches (8 valid, 8 zero-padded), for q-rows 2w, 2w+1.
    v16bf a2[2][2];
#pragma unroll
    for (int j = 0; j < 2; ++j) {
        const int qloc = 2 * w + j;
#pragma unroll
        for (int c = 0; c < 2; ++c) {
            if (nn < 8) frag_from_bf16(a2[j][c], &sQ[nn][qloc][c * 32], h);
            else        frag_zero(a2[j][c]);
        }
    }

    const v8f vzero = {};
    v8f o1[4];
    v8f o2[2][4];
#pragma unroll
    for (int i = 0; i < 4; ++i) o1[i] = vzero;
#pragma unroll
    for (int j = 0; j < 2; ++j)
#pragma unroll
        for (int i = 0; i < 4; ++i) o2[j][i] = vzero;

    float ml[8], ll[8];
#pragma unroll
    for (int r = 0; r < 8; ++r) { ml[r] = -1e30f; ll[r] = 0.0f; }

    for (int t = 0; t < NTILES; ++t) {
        const int kt = k0 + t * KT;

        // Prefetch next tile of the streaming bias (global_prefetch_b8).
        if (t + 1 < NTILES) {
            __builtin_prefetch(kbias + ((size_t)(q0 + 2 * w) * L_ + kt + KT + nn) * D_, 0, 1);
            __builtin_prefetch(vbias + ((size_t)(q0 + 2 * w) * L_ + kt + KT + nn) * D_, 0, 1);
        }

        // ---- attn_1: S1 = Q K^T (batch w) ----
        v8f cs[2];
#pragma unroll
        for (int ntl = 0; ntl < 2; ++ntl) {
            v16bf b0, b1;
            const __bf16* kp = Kbf + ((size_t)w * L_ + kt + ntl * 16 + nn) * D_;
            frag_from_bf16(b0, kp, h);
            frag_from_bf16(b1, kp + 32, h);
            v8f z = wmma_bf16(a1[0], b0, vzero);
            cs[ntl] = wmma_bf16(a1[1], b1, z);
        }

        // ---- attn_2: S2[b,k] = q[:,qi,:] . k_bias[qi,k,:]^T ----
#pragma unroll
        for (int j = 0; j < 2; ++j) {
            const int qloc = 2 * w + j;
            const int qi   = q0 + qloc;
#pragma unroll
            for (int ntl = 0; ntl < 2; ++ntl) {
                const float* bp =
                    kbias + ((size_t)qi * L_ + kt + ntl * 16 + nn) * D_;
                v16bf b0, b1;
                frag_from_f32(b0, bp, h);
                frag_from_f32(b1, bp + 32, h);
                v8f z = wmma_bf16(a2[j][0], b0, vzero);
                z     = wmma_bf16(a2[j][1], b1, z);
#pragma unroll
                for (int r = 0; r < 8; ++r)
                    if (h == 0) sS2[qloc][r][ntl * 16 + nn] = z[r];
            }
        }
        __syncthreads();

        // ---- combine + mask + online softmax (wave w == batch w) ----
        const int mk0 = mask[w * L_ + kt + nn];
        const int mk1 = mask[w * L_ + kt + 16 + nn];
        float scale[8];
#pragma unroll
        for (int r = 0; r < 8; ++r) {
            const int qloc = r + h * 8;
            float s0 = (cs[0][r] + sS2[qloc][w][nn])      * 0.125f;
            float s1 = (cs[1][r] + sS2[qloc][w][16 + nn]) * 0.125f;
            if (mk0 == 0) s0 = -1e9f;
            if (mk1 == 0) s1 = -1e9f;
            float mx = fmaxf(s0, s1);
#pragma unroll
            for (int off = 8; off; off >>= 1)
                mx = fmaxf(mx, __shfl_xor(mx, off, 32));
            const float mnew = fmaxf(ml[r], mx);
            const float sc   = __expf(ml[r] - mnew);
            const float e0   = __expf(s0 - mnew);
            const float e1   = __expf(s1 - mnew);
            float rs = e0 + e1;
#pragma unroll
            for (int off = 8; off; off >>= 1) rs += __shfl_xor(rs, off, 32);
            ll[r] = ll[r] * sc + rs;
            ml[r] = mnew;
            scale[r] = sc;
            sP[w][qloc][nn]      = (__bf16)e0;
            sP[w][qloc][16 + nn] = (__bf16)e1;
            if (nn == 0) sScale[w][qloc] = sc;
        }
#pragma unroll
        for (int n2 = 0; n2 < 4; ++n2)
#pragma unroll
            for (int r = 0; r < 8; ++r) o1[n2][r] *= scale[r];
        __syncthreads();

        // ---- values_1: O1 += P V (batch w); Vt gives contiguous B-frags ----
        v16bf ap;
        frag_from_bf16(ap, &sP[w][nn][0], h);
#pragma unroll
        for (int n2 = 0; n2 < 4; ++n2) {
            v16bf bv_;
            const int d = n2 * 16 + nn;
            frag_from_bf16(bv_, Vt + ((size_t)w * D_ + d) * L_ + kt, h);
            o1[n2] = wmma_bf16(ap, bv_, o1[n2]);
        }

        // ---- values_2: O2 += P_allbatches . v_bias[qi] ----
#pragma unroll
        for (int j = 0; j < 2; ++j) {
            const int qloc = 2 * w + j;
            const int qi   = q0 + qloc;
            float sc2[8];
#pragma unroll
            for (int r = 0; r < 8; ++r) {
                const int m = r + h * 8;
                sc2[r] = (m < 8) ? sScale[m][qloc] : 0.0f;
            }
#pragma unroll
            for (int n2 = 0; n2 < 4; ++n2)
#pragma unroll
                for (int r = 0; r < 8; ++r) o2[j][n2][r] *= sc2[r];

            v16bf ap2;
            if (nn < 8) frag_from_bf16(ap2, &sP[nn][qloc][0], h);
            else        frag_zero(ap2);
#pragma unroll
            for (int n2 = 0; n2 < 4; ++n2) {
                v16bf bb2;
                const int d = n2 * 16 + nn;
                const float* vp = vbias + ((size_t)qi * L_ + kt) * D_ + d;
#pragma unroll
                for (int e = 0; e < 16; ++e)
                    bb2[e] = (__bf16)vp[(size_t)kmap(h, e) * D_];
                o2[j][n2] = wmma_bf16(ap2, bb2, o2[j][n2]);
            }
        }
        __syncthreads();
    }

    // ---- write partials + flash stats ----
    {
        float* dst = O1p + (((size_t)qt * KSPLIT + ks) * B_ + w) * QT * D_;
#pragma unroll
        for (int n2 = 0; n2 < 4; ++n2)
#pragma unroll
            for (int r = 0; r < 8; ++r)
                dst[(r + h * 8) * D_ + n2 * 16 + nn] = o1[n2][r];
        if (nn == 0) {
            float* st = stats + (((size_t)qt * KSPLIT + ks) * B_ + w) * QT * 2;
#pragma unroll
            for (int r = 0; r < 8; ++r) {
                st[(r + h * 8) * 2 + 0] = ml[r];
                st[(r + h * 8) * 2 + 1] = ll[r];
            }
        }
    }
#pragma unroll
    for (int j = 0; j < 2; ++j) {
        const int qloc = 2 * w + j;
        float* dst = O2p + (((size_t)qt * KSPLIT + ks) * QT + qloc) * B_ * D_;
#pragma unroll
        for (int n2 = 0; n2 < 4; ++n2)
#pragma unroll
            for (int r = 0; r < 8; ++r) {
                const int m = r + h * 8;
                if (m < 8) dst[m * D_ + n2 * 16 + nn] = o2[j][n2][r];
            }
    }
}

// ---------------------------------------------------------------------------
// Kernel 3: merge the KSPLIT flash partials (log-sum-exp combine), emit f32.
// ---------------------------------------------------------------------------
__global__ __launch_bounds__(64) void reduce_kernel(
    const float* __restrict__ O1p, const float* __restrict__ O2p,
    const float* __restrict__ stats, float* __restrict__ out)
{
    const int bq   = blockIdx.x;        // b*L + q
    const int b    = bq >> 10;
    const int q    = bq & (L_ - 1);
    const int qt   = q >> 4;
    const int qloc = q & 15;
    const int d    = threadIdx.x;

    float m[KSPLIT], l[KSPLIT], M = -1e30f;
#pragma unroll
    for (int s = 0; s < KSPLIT; ++s) {
        const float* st =
            stats + ((((size_t)qt * KSPLIT + s) * B_ + b) * QT + qloc) * 2;
        m[s] = st[0];
        l[s] = st[1];
        M = fmaxf(M, m[s]);
    }
    float acc = 0.0f, lt = 0.0f;
#pragma unroll
    for (int s = 0; s < KSPLIT; ++s) {
        const float f = __expf(m[s] - M);
        lt += l[s] * f;
        const float v1 =
            O1p[((((size_t)qt * KSPLIT + s) * B_ + b) * QT + qloc) * D_ + d];
        const float v2 =
            O2p[((((size_t)qt * KSPLIT + s) * QT + qloc) * B_ + b) * D_ + d];
        acc += (v1 + v2) * f;
    }
    out[((size_t)b * L_ + q) * D_ + d] = acc / lt;
}

// ---------------------------------------------------------------------------
extern "C" void kernel_launch(void* const* d_in, const int* in_sizes, int n_in,
                              void* d_out, int out_size, void* d_ws, size_t ws_size,
                              hipStream_t stream)
{
    const float* query = (const float*)d_in[0];
    const float* key   = (const float*)d_in[1];
    const float* value = (const float*)d_in[2];
    const float* kbias = (const float*)d_in[3];
    const float* vbias = (const float*)d_in[4];
    const float* Wq    = (const float*)d_in[5];
    const float* bq    = (const float*)d_in[6];
    const float* Wk    = (const float*)d_in[7];
    const float* bk    = (const float*)d_in[8];
    const float* Wv    = (const float*)d_in[9];
    const float* bv    = (const float*)d_in[10];
    const int*   mask  = (const int*)d_in[11];

    char* ws = (char*)d_ws;
    // Workspace map (~20.2 MB):
    __bf16* Qbf  = (__bf16*)(ws + (size_t)0);           // 1 MB
    __bf16* Kbf  = (__bf16*)(ws + ((size_t)1 << 20));   // 1 MB
    __bf16* Vt   = (__bf16*)(ws + ((size_t)2 << 20));   // 1 MB (transposed V)
    __bf16* Wt   = (__bf16*)(ws + ((size_t)3 << 20));   // 384 KB
    float*  O1p  = (float*) (ws + ((size_t)4 << 20));   // 8 MB
    float*  O2p  = (float*) (ws + ((size_t)12 << 20));  // 8 MB
    float*  stats= (float*) (ws + ((size_t)20 << 20));  // 256 KB

    wt_kernel<<<3 * D_, 256, 0, stream>>>(Wq, Wk, Wv, Wt);
    proj_kernel<<<dim3(512, 4, 3), 32, 0, stream>>>(
        query, key, value, Wt, bq, bk, bv, Qbf, Kbf, Vt);
    attn_kernel<<<dim3(L_ / QT, KSPLIT), 256, 0, stream>>>(
        Qbf, Kbf, Vt, kbias, vbias, mask, O1p, O2p, stats);
    reduce_kernel<<<B_ * L_, D_, 0, stream>>>(O1p, O2p, stats, (float*)d_out);
}
